// bidirectionalRNN_26551487824696
// MI455X (gfx1250) — matrix operand order repaired and checked by reading, not compiled, not run on hardware
//
#include <hip/hip_runtime.h>
#include <stdint.h>

// Bidirectional 2-layer LSTM for MI455X (gfx1250), persistent-RNN design.
// - Batched input GEMMs (x @ W_ih^T) via v_wmma_f32_16x16x32_bf16, f32 accum,
//   2m x 4n tiles per wave with register double-buffering of A/B fragments.
// - Gate buffer G stored in WMMA-fragment-blocked layout (16x16 tile = 256
//   contiguous f32), so producer stores and consumer loads are b128-vector.
// - Grid-resident recurrent kernel: 64 blocks (32/direction), W_hh cached in
//   LDS (64KB/block of the 320KB WGP LDS), device-wide atomic barrier per step.
// Workspace requirement: ~1.30 GB (2x 512MB f32 gate buffers + bf16 staging).

#define T_SEQ 1024
#define BATCH 64
#define HID   512
#define G4H   2048   // 4*HID

typedef __attribute__((ext_vector_type(16))) __bf16 v16bf;
typedef __attribute__((ext_vector_type(8)))  __bf16 v8bf;
typedef __attribute__((ext_vector_type(8)))  float  v8f;

// ---------------------------------------------------------------- helpers ---

__device__ __forceinline__ unsigned short f32_to_bf16(float f) {
  unsigned u = __float_as_uint(f);
  u += 0x7FFFu + ((u >> 16) & 1u);          // round to nearest even
  return (unsigned short)(u >> 16);
}

__device__ __forceinline__ float sigmoidf_fast(float x) {
  return 1.0f / (1.0f + __expf(-x));
}

__device__ __forceinline__ v8f wmma_bf16(v16bf a, v16bf b, v8f c) {
  // v_wmma_f32_16x16x32_bf16: D = A(16x32) * B(32x16) + C(16x16 f32)
  return __builtin_amdgcn_wmma_f32_16x16x32_bf16(false, a, false, b,
                                                 (short)0, c, false, false);
}

// A-matrix (16x32 bf16) from row-major [ld] memory.
// Lane L<16 : row M=L,   K = k0..k0+15
// Lane L>=16: row M=L-16,K = k0+16..k0+31
__device__ __forceinline__ v16bf loadA_g(const unsigned short* X, int m0,
                                         int k0, int ld) {
  int lane = threadIdx.x & 31;
  int r = lane & 15, half = lane >> 4;
  const unsigned short* p = X + (size_t)(m0 + r) * ld + k0 + half * 16;
  v8bf lo = *(const v8bf*)p;          // K k0+half*16 .. +7
  v8bf hi = *(const v8bf*)(p + 8);    // K k0+half*16+8 .. +7
  return __builtin_shufflevector(lo, hi, 0, 1, 2, 3, 4, 5, 6, 7,
                                 8, 9, 10, 11, 12, 13, 14, 15);
}

// B-matrix (32x16 bf16) where B = W^T, W row-major [n, 512].
// Lane c<16 holds column c = W row (n0+c), K = k0..k0+15 (contiguous 32B);
// lane c+16 holds K = k0+16..k0+31.
__device__ __forceinline__ v16bf loadB_g(const unsigned short* W, int n0,
                                         int k0) {
  int lane = threadIdx.x & 31;
  int c = lane & 15, half = lane >> 4;
  const unsigned short* p = W + (size_t)(n0 + c) * HID + k0 + half * 16;
  return *(const v16bf*)p;
}

// B-matrix fragment from LDS-cached W rows (row stride 520 ushorts, padded).
__device__ __forceinline__ v16bf loadB_lds(const unsigned short* Wlds, int g,
                                           int cc, int half, int k0) {
  const unsigned short* bp = &Wlds[(size_t)(g * 16 + cc) * 520 + k0 + half * 16];
  v8bf lo = *(const v8bf*)bp;
  v8bf hi = *(const v8bf*)(bp + 8);
  return __builtin_shufflevector(lo, hi, 0, 1, 2, 3, 4, 5, 6, 7,
                                 8, 9, 10, 11, 12, 13, 14, 15);
}

// G is stored fragment-blocked: tile (tileM, tileN) of the [65536, 2048]
// gate matrix lives at G + (tileM*128 + tileN)*256 + lane*8, 8 f32 per lane,
// exactly the v_wmma f32 C/D register layout.
__device__ __forceinline__ float* gfrag(float* G, int tileM, int tileN) {
  return G + (((size_t)tileM * 128) + tileN) * 256 + (threadIdx.x & 31) * 8;
}
__device__ __forceinline__ const float* gfrag_c(const float* G, int tileM,
                                                int tileN) {
  return G + (((size_t)tileM * 128) + tileN) * 256 + (threadIdx.x & 31) * 8;
}

// ---------------------------------------------------------------- kernels ---

__global__ void cvt_f32_bf16(const float* __restrict__ s,
                             unsigned short* __restrict__ d, int n) {
  int i = (blockIdx.x * blockDim.x + threadIdx.x) * 4;
  if (i + 3 < n) {
    float4 v = *(const float4*)(s + i);
    ushort4 o;
    o.x = f32_to_bf16(v.x);
    o.y = f32_to_bf16(v.y);
    o.z = f32_to_bf16(v.z);
    o.w = f32_to_bf16(v.w);
    *(ushort4*)(d + i) = o;
  } else {
    for (; i < n; ++i) d[i] = f32_to_bf16(s[i]);
  }
}

// G(blocked)[4096 x 128 tiles] = X[65536, 512] @ W[2048, 512]^T (bf16->f32).
// One wave: 2 m-tiles x 4 n-tiles, register double-buffered over K so the
// next iteration's global loads overlap the current 8 WMMAs.
__global__ void __launch_bounds__(256) gemm_xw_bf16(
    const unsigned short* __restrict__ X, const unsigned short* __restrict__ W,
    float* __restrict__ G) {
  int wid = blockIdx.x * (blockDim.x >> 5) + (threadIdx.x >> 5);
  int mg = wid >> 5;                  // 0..2047 -> rows mg*32 .. mg*32+31
  int nt = (wid & 31) * 4;            // n-tile base, 0..124
  int n0 = nt * 16;
  int m0 = mg * 32;

  v8f acc[8];
#pragma unroll
  for (int i = 0; i < 8; ++i) acc[i] = (v8f){};

  // preload k=0 fragments
  v16bf A0 = loadA_g(X, m0, 0, HID);
  v16bf A1 = loadA_g(X, m0 + 16, 0, HID);
  v16bf B0 = loadB_g(W, n0 + 0, 0);
  v16bf B1 = loadB_g(W, n0 + 16, 0);
  v16bf B2 = loadB_g(W, n0 + 32, 0);
  v16bf B3 = loadB_g(W, n0 + 48, 0);

#pragma unroll
  for (int kn = 32; kn <= HID; kn += 32) {
    v16bf nA0 = {}, nA1 = {}, nB0 = {}, nB1 = {}, nB2 = {}, nB3 = {};
    if (kn < HID) {   // constant-folded per unrolled iteration
      nA0 = loadA_g(X, m0, kn, HID);
      nA1 = loadA_g(X, m0 + 16, kn, HID);
      nB0 = loadB_g(W, n0 + 0, kn);
      nB1 = loadB_g(W, n0 + 16, kn);
      nB2 = loadB_g(W, n0 + 32, kn);
      nB3 = loadB_g(W, n0 + 48, kn);
    }
    acc[0] = wmma_bf16(A0, B0, acc[0]);
    acc[1] = wmma_bf16(A0, B1, acc[1]);
    acc[2] = wmma_bf16(A0, B2, acc[2]);
    acc[3] = wmma_bf16(A0, B3, acc[3]);
    acc[4] = wmma_bf16(A1, B0, acc[4]);
    acc[5] = wmma_bf16(A1, B1, acc[5]);
    acc[6] = wmma_bf16(A1, B2, acc[6]);
    acc[7] = wmma_bf16(A1, B3, acc[7]);
    A0 = nA0; A1 = nA1;
    B0 = nB0; B1 = nB1; B2 = nB2; B3 = nB3;
  }

#pragma unroll
  for (int j = 0; j < 4; ++j) {
    *(v8f*)gfrag(G, mg * 2 + 0, nt + j) = acc[j];
    *(v8f*)gfrag(G, mg * 2 + 1, nt + j) = acc[4 + j];
  }
}

struct ChainArgs {
  const float* G;              // fragment-blocked [4096 mtile][128 ntile][256]
  const unsigned short* Whh;   // [4H, 512] bf16
  unsigned short* hp0;         // h state ping [B,512] bf16
  unsigned short* hp1;         // h state pong
  float* cstate;               // [B,512] f32 (in-place, owned per-wave)
  unsigned short* hseq;        // [T,B,512] bf16, layer0 only (else null)
  float* outp;                 // d_out slice (ld 1024), layer1 only (else null)
  float* hidH;                 // final-h slot in d_out (ld 1024)
  float* hidC;                 // final-c slot in d_out (ld 1024)
  int grev, orev, hid_step;    // gate-index reversal, out-index reversal
};

// Persistent recurrent kernel: grid = 64 blocks x 128 threads.
// Blocks 0-31: chain A (fwd), 32-63: chain B (bwd). Block jb owns hidden
// columns [jb*16, jb*16+16) across all 4 gates and all 64 batch rows; its
// 64 W_hh rows (64KB bf16) are staged into LDS once.
__global__ void __launch_bounds__(128) lstm_chain(ChainArgs ca, ChainArgs cb,
                                                  unsigned* __restrict__ bar) {
  extern __shared__ unsigned short Wlds[];   // [64][520] padded rows
  const ChainArgs A = (blockIdx.x < 32) ? ca : cb;
  const int jb = blockIdx.x & 31;
  const int col0 = jb * 16;

  // ---- stage W_hh rows for this column slice into LDS (16B chunks) ----
  for (int idx = threadIdx.x; idx < 64 * 64; idx += blockDim.x) {
    int lr = idx >> 6;   // local row 0..63 : gate g = lr>>4, col r = lr&15
    int ch = idx & 63;   // 16-byte chunk within the 1KB row
    int g = lr >> 4, r = lr & 15;
    const uint4* src =
        (const uint4*)(A.Whh + (size_t)(g * HID + col0 + r) * HID) + ch;
    *(uint4*)(&Wlds[(size_t)lr * 520 + ch * 8]) = *src;
  }
  __syncthreads();

  const int wv = threadIdx.x >> 5;   // wave 0..3 -> batch m-tile
  const int m0 = wv * 16;
  const int lane = threadIdx.x & 31;
  const int cc = lane & 15, half = lane >> 4;

  for (int step = 0; step < T_SEQ; ++step) {
    const int tg = A.grev ? (T_SEQ - 1 - step) : step;
    const int to = A.orev ? (T_SEQ - 1 - step) : step;
    const unsigned short* hin = (step & 1) ? A.hp1 : A.hp0;
    unsigned short* hout = (step & 1) ? A.hp0 : A.hp1;
    const int tileM = tg * 4 + wv;

    // init accumulators with precomputed input gates (i,f,g,o sections);
    // fragment-blocked -> one contiguous 32B load per gate tile
    v8f acc[4];
#pragma unroll
    for (int g = 0; g < 4; ++g)
      acc[g] = *(const v8f*)gfrag_c(A.G, tileM, g * 32 + jb);

    if (step + 1 < T_SEQ) {   // prefetch next step's 4 gate fragments
      int tgn = A.grev ? (T_SEQ - 2 - step) : (step + 1);
      int tMn = tgn * 4 + wv;
#pragma unroll
      for (int g = 0; g < 4; ++g)
        __builtin_prefetch(gfrag_c(A.G, tMn, g * 32 + jb), 0, 1);
    }

    // gates += h @ W_hh^T  (B operands from LDS; load all 4 before the WMMAs
    // so DS waits overlap matrix math)
#pragma unroll 4
    for (int k0 = 0; k0 < HID; k0 += 32) {
      v16bf a = loadA_g(hin, m0, k0, HID);
      v16bf b0 = loadB_lds(Wlds, 0, cc, half, k0);
      v16bf b1 = loadB_lds(Wlds, 1, cc, half, k0);
      v16bf b2 = loadB_lds(Wlds, 2, cc, half, k0);
      v16bf b3 = loadB_lds(Wlds, 3, cc, half, k0);
      acc[0] = wmma_bf16(a, b0, acc[0]);
      acc[1] = wmma_bf16(a, b1, acc[1]);
      acc[2] = wmma_bf16(a, b2, acc[2]);
      acc[3] = wmma_bf16(a, b3, acc[3]);
    }

    // fused LSTM cell elementwise on this wave's (16 batch x 16 col) patch
#pragma unroll
    for (int r = 0; r < 8; ++r) {
      int brow = m0 + half * 8 + r;
      int col = col0 + cc;
      float iv = sigmoidf_fast(acc[0][r]);
      float fv = sigmoidf_fast(acc[1][r]);
      float gv = tanhf(acc[2][r]);
      float ov = sigmoidf_fast(acc[3][r]);
      size_t cidx = (size_t)brow * HID + col;
      float cn = fv * A.cstate[cidx] + iv * gv;
      A.cstate[cidx] = cn;
      float hn = ov * tanhf(cn);
      unsigned short hb = f32_to_bf16(hn);
      hout[cidx] = hb;
      if (A.hseq) A.hseq[(size_t)step * (BATCH * HID) + cidx] = hb;
      if (A.outp)
        A.outp[(size_t)to * (BATCH * 2 * HID) + (size_t)brow * (2 * HID) +
               col] = hn;
      if (step == A.hid_step) {
        A.hidH[(size_t)brow * (2 * HID) + col] = hn;
        A.hidC[(size_t)brow * (2 * HID) + col] = cn;
      }
    }

    // ---- device-wide barrier (monotonic counter, 64 blocks) ----
    __syncthreads();
    if (threadIdx.x == 0) {
      __hip_atomic_fetch_add(bar, 1u, __ATOMIC_RELEASE,
                             __HIP_MEMORY_SCOPE_AGENT);
      unsigned tgt = (unsigned)(step + 1) * 64u;
      while (__hip_atomic_load(bar, __ATOMIC_ACQUIRE,
                               __HIP_MEMORY_SCOPE_AGENT) < tgt)
        __builtin_amdgcn_s_sleep(1);
    }
    __syncthreads();
    __threadfence();   // acquire: make other blocks' h stores visible
  }
}

// ------------------------------------------------------------------ host ---

extern "C" void kernel_launch(void* const* d_in, const int* in_sizes, int n_in,
                              void* d_out, int out_size, void* d_ws,
                              size_t ws_size, hipStream_t stream) {
  (void)in_sizes; (void)n_in; (void)out_size; (void)ws_size;
  const float* x = (const float*)d_in[0];
  // order: w_ih_f0, w_hh_f0, w_ih_f1, w_hh_f1, w_ih_b0, w_hh_b0, w_ih_b1, w_hh_b1
  const float* wf32[8];
  for (int i = 0; i < 8; ++i) wf32[i] = (const float*)d_in[1 + i];

  char* ws = (char*)d_ws;
  size_t off = 0;
  auto take = [&](size_t bytes) -> char* {
    char* p = ws + off;
    off += (bytes + 255) & ~(size_t)255;
    return p;
  };

  unsigned short* xbf = (unsigned short*)take((size_t)T_SEQ * BATCH * HID * 2);
  unsigned short* wbf[8];
  for (int i = 0; i < 8; ++i)
    wbf[i] = (unsigned short*)take((size_t)G4H * HID * 2);
  float* Gf = (float*)take((size_t)T_SEQ * BATCH * G4H * 4);
  float* Gb = (float*)take((size_t)T_SEQ * BATCH * G4H * 4);
  unsigned short* hseqf = (unsigned short*)take((size_t)T_SEQ * BATCH * HID * 2);
  unsigned short* hseqb = (unsigned short*)take((size_t)T_SEQ * BATCH * HID * 2);
  unsigned short* hpA0 = (unsigned short*)take((size_t)BATCH * HID * 2);
  unsigned short* hpA1 = (unsigned short*)take((size_t)BATCH * HID * 2);
  unsigned short* hpB0 = (unsigned short*)take((size_t)BATCH * HID * 2);
  unsigned short* hpB1 = (unsigned short*)take((size_t)BATCH * HID * 2);
  float* cA = (float*)take((size_t)BATCH * HID * 4);
  float* cB = (float*)take((size_t)BATCH * HID * 4);
  unsigned* bar = (unsigned*)take(256);

  float* out_output = (float*)d_out;                               // [T,B,1024]
  float* out_h = out_output + (size_t)T_SEQ * BATCH * 2 * HID;     // [2,B,1024]
  float* out_c = out_h + (size_t)2 * BATCH * 2 * HID;              // [2,B,1024]

  // ---- fp32 -> bf16 staging ----
  {
    int n = T_SEQ * BATCH * HID;
    cvt_f32_bf16<<<(n / 4 + 255) / 256, 256, 0, stream>>>(x, xbf, n);
  }
  for (int i = 0; i < 8; ++i) {
    int n = G4H * HID;
    cvt_f32_bf16<<<(n / 4 + 255) / 256, 256, 0, stream>>>(wf32[i], wbf[i], n);
  }

  const int gemmBlocks = (2048 * 32) / 8;   // 65536 waves, 8 waves/block
  const size_t ldsBytes = 64 * 520 * 2;     // 66560 B < 320KB WGP LDS
  const size_t stBytes = (size_t)BATCH * HID;

  // =================== Phase 0: layer 0, fwd + bwd ===================
  gemm_xw_bf16<<<gemmBlocks, 256, 0, stream>>>(xbf, wbf[0], Gf);  // w_ih_f0
  gemm_xw_bf16<<<gemmBlocks, 256, 0, stream>>>(xbf, wbf[4], Gb);  // w_ih_b0
  hipMemsetAsync(hpA0, 0, stBytes * 2, stream);
  hipMemsetAsync(hpA1, 0, stBytes * 2, stream);
  hipMemsetAsync(hpB0, 0, stBytes * 2, stream);
  hipMemsetAsync(hpB1, 0, stBytes * 2, stream);
  hipMemsetAsync(cA, 0, stBytes * 4, stream);
  hipMemsetAsync(cB, 0, stBytes * 4, stream);
  hipMemsetAsync(bar, 0, sizeof(unsigned), stream);
  {
    ChainArgs f0 = {Gf, wbf[1], hpA0, hpA1, cA, hseqf, nullptr,
                    out_h + 0, out_c + 0, 0, 0, T_SEQ - 1};
    ChainArgs b0 = {Gb, wbf[5], hpB0, hpB1, cB, hseqb, nullptr,
                    out_h + HID, out_c + HID, 1, 0, 0};
    lstm_chain<<<64, 128, ldsBytes, stream>>>(f0, b0, bar);
  }

  // =================== Phase 1: layer 1, fwd + bwd ===================
  gemm_xw_bf16<<<gemmBlocks, 256, 0, stream>>>(hseqf, wbf[2], Gf);  // w_ih_f1
  gemm_xw_bf16<<<gemmBlocks, 256, 0, stream>>>(hseqb, wbf[6], Gb);  // w_ih_b1
  hipMemsetAsync(hpA0, 0, stBytes * 2, stream);
  hipMemsetAsync(hpA1, 0, stBytes * 2, stream);
  hipMemsetAsync(hpB0, 0, stBytes * 2, stream);
  hipMemsetAsync(hpB1, 0, stBytes * 2, stream);
  hipMemsetAsync(cA, 0, stBytes * 4, stream);
  hipMemsetAsync(cB, 0, stBytes * 4, stream);
  hipMemsetAsync(bar, 0, sizeof(unsigned), stream);
  {
    const size_t l1 = (size_t)BATCH * 2 * HID;   // layer-1 offset in h/c
    ChainArgs f1 = {Gf, wbf[3], hpA0, hpA1, cA, nullptr, out_output + 0,
                    out_h + l1 + 0, out_c + l1 + 0, 0, 0, T_SEQ - 1};
    ChainArgs b1 = {Gb, wbf[7], hpB0, hpB1, cB, nullptr, out_output + HID,
                    out_h + l1 + HID, out_c + l1 + HID, 0, 1, 0};
    lstm_chain<<<64, 128, ldsBytes, stream>>>(f1, b1, bar);
  }
}
